// MultiHeadAttention_19808389169485
// MI455X (gfx1250) — matrix operand order repaired and checked by reading, not compiled, not run on hardware
//
#include <hip/hip_runtime.h>
#include <math.h>

typedef _Float16 h16;
typedef __attribute__((ext_vector_type(16))) _Float16 v16h;
typedef __attribute__((ext_vector_type(8)))  _Float16 v8h;
typedef __attribute__((ext_vector_type(8)))  float    v8f;
typedef __attribute__((ext_vector_type(4)))  int      v4i;

#define DIM 2048
#define SEQ 2048
#define NH  16
#define DH  128

#if defined(__HIP_DEVICE_COMPILE__) && defined(__gfx1250__) && \
    __has_builtin(__builtin_amdgcn_global_load_async_to_lds_b128)
#define HAS_ASYNC_LDS 1
typedef __attribute__((address_space(1))) v4i g_v4i;
typedef __attribute__((address_space(3))) v4i l_v4i;
#else
#define HAS_ASYNC_LDS 0
#endif

#define WMMA_F16(a, b, c) \
  __builtin_amdgcn_wmma_f32_16x16x32_f16(false, (a), false, (b), (short)0, (c), false, false)

__device__ __forceinline__ v16h pack16(v8h lo, v8h hi) {
  v16h r;
#pragma unroll
  for (int i = 0; i < 8; i++) { r[i] = lo[i]; r[i + 8] = hi[i]; }
  return r;
}

__device__ __forceinline__ v8h cvt8(float4 u, float4 v) {
  v8h r;
  r[0] = (h16)u.x; r[1] = (h16)u.y; r[2] = (h16)u.z; r[3] = (h16)u.w;
  r[4] = (h16)v.x; r[5] = (h16)v.y; r[6] = (h16)v.z; r[7] = (h16)v.w;
  return r;
}

// ---------------------------------------------------------------------------
// Generic projection GEMM:  C[M=4096, N=2048] = A[M,K] @ W[N,K]^T + bias
// MODE 0: plain f32 [M,N] ; MODE 1: head-split f32 [bs,H,S,dh] ;
// MODE 2: head-split f16 [bs,H,S,dh]
// 256 threads (8 waves); 128x128 C tile; K staged 32/step, double-buffered LDS.
// ---------------------------------------------------------------------------
template <int MODE>
__global__ void __launch_bounds__(256)
proj_kernel(const float* __restrict__ A, const float* __restrict__ W,
            const float* __restrict__ bias, float* __restrict__ outF,
            h16* __restrict__ outH) {
  const int K = DIM, N = DIM;
  __shared__ __align__(16) h16 ldsA[2][128 * 40];
  __shared__ __align__(16) h16 ldsB[2][128 * 40];

  const int tid = threadIdx.x;
  const int lane = tid & 31, wv = tid >> 5;
  const int sel = lane >> 4, l16 = lane & 15;
  const int m0 = blockIdx.y * 128, n0 = blockIdx.x * 128;

  v8f acc[8];
#pragma unroll
  for (int t = 0; t < 8; t++)
#pragma unroll
    for (int i = 0; i < 8; i++) acc[t][i] = 0.0f;

  const int srow = tid >> 1, shalf = tid & 1;
  const float* gA = A + (size_t)(m0 + srow) * K + 16 * shalf;
  const float* gB = W + (size_t)(n0 + srow) * K + 16 * shalf;

  // Preload first K-chunk into registers.
  float4 fa[4], fb[4];
#pragma unroll
  for (int c = 0; c < 4; c++) {
    fa[c] = *(const float4*)(gA + 4 * c);
    fb[c] = *(const float4*)(gB + 4 * c);
  }

  for (int k0 = 0; k0 < K; k0 += 32) {
    const int buf = (k0 >> 5) & 1;
    // Commit staged registers to this iteration's LDS buffer.
    {
      h16* sA = &ldsA[buf][srow * 40 + 16 * shalf];
      h16* sB = &ldsB[buf][srow * 40 + 16 * shalf];
      *(v8h*)(sA)     = cvt8(fa[0], fa[1]);
      *(v8h*)(sA + 8) = cvt8(fa[2], fa[3]);
      *(v8h*)(sB)     = cvt8(fb[0], fb[1]);
      *(v8h*)(sB + 8) = cvt8(fb[2], fb[3]);
    }
    // Kick off next chunk's global loads (overlap with WMMA below) and
    // prefetch the chunk after that toward L2.
    if (k0 + 32 < K) {
#pragma unroll
      for (int c = 0; c < 4; c++) {
        fa[c] = *(const float4*)(gA + k0 + 32 + 4 * c);
        fb[c] = *(const float4*)(gB + k0 + 32 + 4 * c);
      }
      __builtin_prefetch(gA + k0 + 64, 0, 3);
      __builtin_prefetch(gB + k0 + 64, 0, 3);
    }
    __syncthreads();

    // Bulk-load all fragments, then issue 8 back-to-back WMMAs.
    v16h af;
    {
      const int r = wv * 16 + l16;
      af = pack16(*(const v8h*)&ldsA[buf][r * 40 + 8 * sel],
                  *(const v8h*)&ldsA[buf][r * 40 + 8 * sel + 16]);
    }
    v16h bf[8];
#pragma unroll
    for (int t = 0; t < 8; t++) {
      const int r = t * 16 + l16;
      bf[t] = pack16(*(const v8h*)&ldsB[buf][r * 40 + 8 * sel],
                     *(const v8h*)&ldsB[buf][r * 40 + 8 * sel + 16]);
    }
#pragma unroll
    for (int t = 0; t < 8; t++) acc[t] = WMMA_F16(af, bf[t], acc[t]);
  }

#pragma unroll
  for (int t = 0; t < 8; t++) {
    const int n = n0 + t * 16 + l16;
    const float bv = bias[n];
#pragma unroll
    for (int r = 0; r < 8; r++) {
      const int m = m0 + wv * 16 + r + (sel << 3);
      const float val = acc[t][r] + bv;
      if (MODE == 0) {
        outF[(size_t)m * N + n] = val;
      } else {
        const int b = m >> 11, s = m & 2047, h = n >> 7, d = n & 127;
        const size_t idx = (((size_t)(b * NH + h)) * SEQ + s) * DH + d;
        if (MODE == 1) outF[idx] = val;
        else           outH[idx] = (h16)val;
      }
    }
  }
}

// ---------------------------------------------------------------------------
// L2-normalize each dh=128 row, optional attention_scale, emit f16.
// One wave per row, 8 rows per block.
// ---------------------------------------------------------------------------
__global__ void __launch_bounds__(256)
norm_kernel(const float* __restrict__ in, h16* __restrict__ out,
            const float* __restrict__ scale_ptr, int use_scale) {
  const int lane = threadIdx.x & 31, wv = threadIdx.x >> 5;
  const size_t row = (size_t)blockIdx.x * 8 + wv;
  const float4 v = *(const float4*)(in + row * DH + lane * 4);
  float s = v.x * v.x + v.y * v.y + v.z * v.z + v.w * v.w;
#pragma unroll
  for (int m = 16; m >= 1; m >>= 1) s += __shfl_xor(s, m, 32);
  const float nrm = fmaxf(sqrtf(s), 1e-12f);
  const float sc = (use_scale ? scale_ptr[0] : 1.0f) / nrm;
  h16* q = out + row * DH + lane * 4;
  q[0] = (h16)(v.x * sc); q[1] = (h16)(v.y * sc);
  q[2] = (h16)(v.z * sc); q[3] = (h16)(v.w * sc);
}

// ---------------------------------------------------------------------------
// Flash attention: block = 128 queries of one (b,h); 8 waves x 16 q-rows.
// Streams 64-key blocks through LDS with online softmax; ctx kept in regs.
// K block staged via GLOBAL_LOAD_ASYNC_TO_LDS (no VGPR round-trip, tracked
// on ASYNCcnt); V (transposed) and mask (int -> additive bias) must pass
// through VGPRs.
// ---------------------------------------------------------------------------
__global__ void __launch_bounds__(256)
attn_kernel(const h16* __restrict__ Q, const h16* __restrict__ Kh,
            const h16* __restrict__ Vh, const int* __restrict__ mask,
            float* __restrict__ ctx) {
  __shared__ __align__(16) h16   ldsK[64 * 136];    // [key][dh], pad-136
  __shared__ __align__(16) h16   ldsV[128 * 72];    // [dh][key], pad-72
  __shared__ __align__(16) float ldsM[128 * 64];    // additive mask bias
  __shared__ __align__(16) h16   ldsP[8 * 16 * 72]; // per-wave P tiles

  const int tid = threadIdx.x, lane = tid & 31, wv = tid >> 5;
  const int sel = lane >> 4, l16 = lane & 15;
  const int q0 = blockIdx.x * 128, hh = blockIdx.y, bb = blockIdx.z;
  const float NEG_INF = -INFINITY;

  const size_t headoff = ((size_t)(bb * NH + hh)) * SEQ * DH;
  const h16* qp = Q + headoff;
  const h16* kp = Kh + headoff;
  const h16* vp = Vh + headoff;
  const int* mp = mask + (size_t)bb * SEQ * SEQ;

  // Preload this wave's 16 q-rows as 4 A-fragments over dh=128 (q pre-scaled).
  v16h qf[4];
#pragma unroll
  for (int kc = 0; kc < 4; kc++) {
    const h16* base = qp + (size_t)(q0 + wv * 16 + l16) * DH + 32 * kc + 8 * sel;
    qf[kc] = pack16(*(const v8h*)base, *(const v8h*)(base + 16));
  }

  v8f acc[8];
#pragma unroll
  for (int t = 0; t < 8; t++)
#pragma unroll
    for (int i = 0; i < 8; i++) acc[t][i] = 0.0f;
  float runm[8], runl[8];
#pragma unroll
  for (int r = 0; r < 8; r++) { runm[r] = NEG_INF; runl[r] = 0.0f; }

  h16* pw = &ldsP[wv * 16 * 72];

  for (int j = 0; j < SEQ / 64; j++) {
    __syncthreads();
    // Stage K block [64 x 128] row-major.
#if HAS_ASYNC_LDS
#pragma unroll
    for (int c = 0; c < 4; c++) {
      const int ch = tid + 256 * c;
      const int key = ch >> 4, off = (ch & 15) * 8;
      __builtin_amdgcn_global_load_async_to_lds_b128(
          (g_v4i*)(void*)(kp + (size_t)(j * 64 + key) * DH + off),
          (l_v4i*)(void*)&ldsK[key * 136 + off],
          0, 0);
    }
#else
#pragma unroll
    for (int c = 0; c < 4; c++) {
      const int ch = tid + 256 * c;
      const int key = ch >> 4, off = (ch & 15) * 8;
      v8h kv = *(const v8h*)(kp + (size_t)(j * 64 + key) * DH + off);
      *(v8h*)&ldsK[key * 136 + off] = kv;
    }
#endif
    // Stage V block transposed: [dh][key].
#pragma unroll
    for (int c = 0; c < 4; c++) {
      const int ch = tid + 256 * c;
      const int key = ch >> 4, off = (ch & 15) * 8;
      v8h vv = *(const v8h*)(vp + (size_t)(j * 64 + key) * DH + off);
#pragma unroll
      for (int i = 0; i < 8; i++) ldsV[(off + i) * 72 + key] = vv[i];
    }
    // Stage mask bias (0 or -inf) for [128 q x 64 k].
    {
      const int row = tid >> 1, half = tid & 1;
      const int* mrow = mp + (size_t)(q0 + row) * SEQ + j * 64 + 32 * half;
#pragma unroll
      for (int c = 0; c < 8; c++) {
        int4 mv = *(const int4*)(mrow + 4 * c);
        float4 bv;
        bv.x = mv.x ? 0.0f : NEG_INF;
        bv.y = mv.y ? 0.0f : NEG_INF;
        bv.z = mv.z ? 0.0f : NEG_INF;
        bv.w = mv.w ? 0.0f : NEG_INF;
        *(float4*)&ldsM[row * 64 + 32 * half + 4 * c] = bv;
      }
    }
    // Pull next key block toward L2 while this one is consumed.
    if (j + 1 < SEQ / 64) {
      const int key = tid >> 2; // 0..63
      __builtin_prefetch(kp + (size_t)((j + 1) * 64 + key) * DH, 0, 3);
      __builtin_prefetch(vp + (size_t)((j + 1) * 64 + key) * DH, 0, 3);
    }
#if HAS_ASYNC_LDS
#if __has_builtin(__builtin_amdgcn_s_wait_asynccnt)
    __builtin_amdgcn_s_wait_asynccnt(0);
#else
    asm volatile("s_wait_asynccnt 0x0" ::: "memory");
#endif
#endif
    __syncthreads();

    // Scores: 16 q-rows x 64 keys. Bulk-load 4 K-frags, then 4 WMMAs.
    v8f sc4[4];
#pragma unroll
    for (int kb = 0; kb < 4; kb++) {
      v8f s;
#pragma unroll
      for (int i = 0; i < 8; i++) s[i] = 0.0f;
      v16h kf[4];
#pragma unroll
      for (int kc = 0; kc < 4; kc++) {
        const h16* base = &ldsK[(kb * 16 + l16) * 136 + 32 * kc + 8 * sel];
        kf[kc] = pack16(*(const v8h*)base, *(const v8h*)(base + 16));
      }
#pragma unroll
      for (int kc = 0; kc < 4; kc++) s = WMMA_F16(qf[kc], kf[kc], s);
#pragma unroll
      for (int r = 0; r < 8; r++) {
        const int ql = wv * 16 + r + (sel << 3);
        s[r] += ldsM[ql * 64 + kb * 16 + l16];
      }
      sc4[kb] = s;
    }

    // Online softmax (row reductions across the 16-lane half).
    float scale[8];
#pragma unroll
    for (int r = 0; r < 8; r++) {
      float t = fmaxf(fmaxf(sc4[0][r], sc4[1][r]), fmaxf(sc4[2][r], sc4[3][r]));
      t = fmaxf(t, __shfl_xor(t, 8, 32));
      t = fmaxf(t, __shfl_xor(t, 4, 32));
      t = fmaxf(t, __shfl_xor(t, 2, 32));
      t = fmaxf(t, __shfl_xor(t, 1, 32));
      const float nm = fmaxf(runm[r], t);
      scale[r] = __expf(runm[r] - nm);
      runm[r] = nm;
    }
#pragma unroll
    for (int t = 0; t < 8; t++)
#pragma unroll
      for (int r = 0; r < 8; r++) acc[t][r] *= scale[r];

#pragma unroll
    for (int r = 0; r < 8; r++) {
      float ps = 0.0f;
#pragma unroll
      for (int kb = 0; kb < 4; kb++) {
        const float pv = __expf(sc4[kb][r] - runm[r]);
        ps += pv;
        pw[(r + (sel << 3)) * 72 + kb * 16 + l16] = (h16)pv;
      }
      ps += __shfl_xor(ps, 8, 32);
      ps += __shfl_xor(ps, 4, 32);
      ps += __shfl_xor(ps, 2, 32);
      ps += __shfl_xor(ps, 1, 32);
      runl[r] = runl[r] * scale[r] + ps;
    }

    // P fragments (A-operand) from this wave's LDS tile.
    v16h pf[2];
#pragma unroll
    for (int pc = 0; pc < 2; pc++) {
      const h16* base = &pw[l16 * 72 + 32 * pc + 8 * sel];
      pf[pc] = pack16(*(const v8h*)base, *(const v8h*)(base + 16));
    }
    // ctx += P @ V : nt-pairs, 4 V-frags bulk-loaded, 4 WMMAs interleaved
    // across two accumulators so consecutive WMMAs don't chain on one D.
#pragma unroll
    for (int np = 0; np < 4; np++) {
      const int nt0 = np * 2, nt1 = np * 2 + 1;
      v16h vf[4];
#pragma unroll
      for (int pc = 0; pc < 2; pc++) {
        const h16* b0 = &ldsV[(nt0 * 16 + l16) * 72 + 32 * pc + 8 * sel];
        const h16* b1 = &ldsV[(nt1 * 16 + l16) * 72 + 32 * pc + 8 * sel];
        vf[pc * 2]     = pack16(*(const v8h*)b0, *(const v8h*)(b0 + 16));
        vf[pc * 2 + 1] = pack16(*(const v8h*)b1, *(const v8h*)(b1 + 16));
      }
      acc[nt0] = WMMA_F16(pf[0], vf[0], acc[nt0]);
      acc[nt1] = WMMA_F16(pf[0], vf[1], acc[nt1]);
      acc[nt0] = WMMA_F16(pf[1], vf[2], acc[nt0]);
      acc[nt1] = WMMA_F16(pf[1], vf[3], acc[nt1]);
    }
  }

  // Epilogue: divide by row sums, write merged-head ctx f32 [bs,S,dim].
#pragma unroll
  for (int nt = 0; nt < 8; nt++) {
#pragma unroll
    for (int r = 0; r < 8; r++) {
      const int s = q0 + wv * 16 + r + (sel << 3);
      const int col = hh * DH + nt * 16 + l16;
      ctx[((size_t)bb * SEQ + s) * DIM + col] = acc[nt][r] / runl[r];
    }
  }
}

// ---------------------------------------------------------------------------
extern "C" void kernel_launch(void* const* d_in, const int* in_sizes, int n_in,
                              void* d_out, int out_size, void* d_ws, size_t ws_size,
                              hipStream_t stream) {
  (void)in_sizes; (void)n_in; (void)out_size; (void)ws_size;
  const float* x   = (const float*)d_in[0];
  const int*   msk = (const int*)d_in[1];
  const float* Wq  = (const float*)d_in[2];
  const float* bq  = (const float*)d_in[3];
  const float* Wk  = (const float*)d_in[4];
  const float* bk  = (const float*)d_in[5];
  const float* Wv  = (const float*)d_in[6];
  const float* bv  = (const float*)d_in[7];
  const float* Wo  = (const float*)d_in[8];
  const float* bo  = (const float*)d_in[9];
  const float* asc = (const float*)d_in[10];
  float* out = (float*)d_out;

  const size_t EL = (size_t)2 * NH * SEQ * DH; // 8,388,608 elements
  char* ws = (char*)d_ws;
  float* qf32 = (float*)ws; ws += EL * 4;      // reused as ctx after norm
  float* kf32 = (float*)ws; ws += EL * 4;
  h16*   q16  = (h16*)ws;   ws += EL * 2;
  h16*   k16  = (h16*)ws;   ws += EL * 2;
  h16*   v16_ = (h16*)ws;   ws += EL * 2;
  float* ctx  = qf32;

  const dim3 blk(256);
  const dim3 gproj(16, 32);

  proj_kernel<1><<<gproj, blk, 0, stream>>>(x, Wq, bq, qf32, nullptr);
  proj_kernel<1><<<gproj, blk, 0, stream>>>(x, Wk, bk, kf32, nullptr);
  proj_kernel<2><<<gproj, blk, 0, stream>>>(x, Wv, bv, nullptr, v16_);
  norm_kernel<<<dim3((2 * NH * SEQ) / 8), blk, 0, stream>>>(qf32, q16, asc, 1);
  norm_kernel<<<dim3((2 * NH * SEQ) / 8), blk, 0, stream>>>(kf32, k16, asc, 0);
  attn_kernel<<<dim3(SEQ / 128, NH, 2), blk, 0, stream>>>(q16, k16, v16_, msk, ctx);
  proj_kernel<0><<<gproj, blk, 0, stream>>>(ctx, Wo, bo, out, nullptr);
}
